// WaveAttentionCorrelation_78056735637605
// MI455X (gfx1250) — matrix operand order repaired and checked by reading, not compiled, not run on hardware
//
#include <hip/hip_runtime.h>

// ---------------- Problem constants (from reference) ----------------
#define DIMC   384          // input channels
#define DIMQ   192          // q/k/v channels
#define NHEAD  8
#define HEADD  24           // DIMQ / NHEAD
#define BATCH  4
#define IMH    256
#define IMW    256
#define NPIX   (IMH*IMW)    // 65536 pixels per batch image
#define HPIX   (NPIX/4)     // 16384 pixels at half resolution (128x128)
#define SCALE_F 0.2041241452319315f   // 24^-0.5

typedef __attribute__((ext_vector_type(16))) __bf16 v16bf;
typedef __attribute__((ext_vector_type(8)))  float  v8f;

static __device__ __forceinline__ __bf16 f2bf(float f) {
  return (__bf16)f;   // native f32->bf16 (v_cvt_*_bf16_f32 class on gfx1250)
}

static __device__ __forceinline__ v8f wmma_bf16(v16bf a, v16bf b, v8f c) {
  // D = A(16x32 bf16) * B(32x16 bf16) + C(16x16 f32)
  return __builtin_amdgcn_wmma_f32_16x16x32_bf16(false, a, false, b,
                                                 (short)0, c, false, false);
}

// K index inside a 32-wide tile for the A-fragment element e of a lane
// (contiguous: lanes 0-15 hold K {0..15}, lanes 16-31 hold {16..31})
static __device__ __forceinline__ int a_kidx(int e, int lane) {
  return e + 16 * (lane >> 4);
}

// ---------------- TDM: DMA a linear byte range global -> LDS ----------------
#if __has_builtin(__builtin_amdgcn_tensor_load_to_lds)
#define HAVE_TDM 1
typedef unsigned int u32x4 __attribute__((ext_vector_type(4)));
typedef int          i32x8 __attribute__((ext_vector_type(8)));
typedef int          i32x4 __attribute__((ext_vector_type(4)));

// 1-D copy: nbytes must be a multiple of 4 and nbytes/4 <= 65535 (tile_dim0).
static __device__ __forceinline__ void tdm_copy_to_lds(void* lds_dst,
                                                       const void* gsrc,
                                                       unsigned nbytes) {
  unsigned ndw = nbytes >> 2;
  unsigned lds_addr = (unsigned)(unsigned long long)(uintptr_t)lds_dst;
  unsigned long long ga = (unsigned long long)(uintptr_t)gsrc;
  u32x4 g0;
  g0.x = 1u;                                    // count=1, user mode, no gather
  g0.y = lds_addr;                              // lds_addr [63:32]
  g0.z = (unsigned)(ga & 0xFFFFFFFFu);          // global_addr lo
  g0.w = (unsigned)((ga >> 32) & 0x01FFFFFFu)   // global_addr [56:32]
       | (2u << 30);                            // type=2 ("image")
  i32x8 g1;
  g1[0] = (int)(2u << 16);                      // data_size=4B, wg_mask=0
  g1[1] = (int)((ndw & 0xFFFFu) << 16);         // tensor_dim0[15:0] @ [63:48]
  g1[2] = (int)(((ndw >> 16) & 0xFFFFu)         // tensor_dim0[31:16]
       | (1u << 16));                           // tensor_dim1 = 1
  g1[3] = (int)((ndw & 0xFFFFu) << 16);         // tile_dim0 = ndw @ [127:112]
  g1[4] = 1;                                    // tile_dim1=1, tile_dim2=0
  g1[5] = (int)ndw;                             // tensor_dim0_stride lo32
  g1[6] = 0;
  g1[7] = 0;
  i32x4 z4 = {0, 0, 0, 0};
  i32x8 z8 = {0, 0, 0, 0, 0, 0, 0, 0};
  // 6-arg form on this toolchain: (g0, g1, g2, g3, extra, cpol)
  __builtin_amdgcn_tensor_load_to_lds(g0, g1, z4, z4, z8, 0);
}
#else
#define HAVE_TDM 0
#endif

// Stage `nbytes` of fragment data into LDS (whole block participates).
static __device__ __forceinline__ void stage_weights(__bf16* lds_dst,
                                                     const __bf16* gsrc,
                                                     unsigned nbytes) {
#if HAVE_TDM
  if (threadIdx.x < 32) {                       // one wave issues the DMA
    tdm_copy_to_lds(lds_dst, gsrc, nbytes);
    __builtin_amdgcn_s_wait_tensorcnt(0);
  }
  __syncthreads();
#else
  const unsigned* s = (const unsigned*)gsrc;
  unsigned* d = (unsigned*)lds_dst;
  for (unsigned i = threadIdx.x; i < (nbytes >> 2); i += blockDim.x) d[i] = s[i];
  __syncthreads();
#endif
}

// ---------------- Weight packing: f32 row-major -> bf16 A-fragments ----------
// Fragment order: frag = mtile*KT + ktile, each frag = 32 lanes x 16 halves.
__global__ void pack_w_kernel(const float* __restrict__ src,
                              __bf16* __restrict__ dst, int M, int K) {
  int tid = blockIdx.x * blockDim.x + threadIdx.x;
  int KT = K >> 5;
  int total = (M >> 4) * KT * 512;
  if (tid >= total) return;
  int frag = tid >> 9;
  int within = tid & 511;
  int lane = within >> 4;
  int e = within & 15;
  int mt = frag / KT, kt = frag % KT;
  int row = mt * 16 + (lane & 15);
  int col = kt * 32 + a_kidx(e, lane);
  dst[tid] = f2bf(src[(size_t)row * K + col]);
}

// ---------------- Channel GEMM: Y[b,m,n] = sum_k W[m,k] * X[b,k,n] (+bias) ----
// Weights for the whole 192-channel M-block are TDM-staged into LDS once;
// the K loop reads A fragments via ds_load and keeps 12 WMMAs back-to-back.
__global__ void gemm_chan_kernel(const __bf16* __restrict__ Apack,
                                 const float* __restrict__ X,
                                 float* __restrict__ Y,
                                 const float* __restrict__ bias,
                                 int M, int K) {
  __shared__ __bf16 aw[12 * 12 * 512];          // up to 144 KB (Q GEMM)
  const int KT = K >> 5;
  const int mBaseTile = blockIdx.y * 12;
  stage_weights(aw, Apack + (size_t)mBaseTile * KT * 512,
                (unsigned)(12 * KT * 512 * 2));

  const int lane = threadIdx.x & 31;
  const int wave = blockIdx.x * (blockDim.x >> 5) + (threadIdx.x >> 5);
  const int ntile = wave & 4095;                // NPIX/16 column tiles
  const int b     = wave >> 12;
  const int col   = lane & 15;
  const int kh    = lane >> 4;
  const int n     = ntile * 16 + col;

  v8f zero = {0.f,0.f,0.f,0.f,0.f,0.f,0.f,0.f};
  v8f acc[12];
#pragma unroll
  for (int i = 0; i < 12; ++i) acc[i] = zero;

  for (int kt = 0; kt < KT; ++kt) {
    const float* xb = X + ((size_t)b * K + (size_t)(kt * 32 + kh * 16)) * NPIX + n;
    if (kt + 1 < KT)
      __builtin_prefetch(xb + (size_t)32 * NPIX, 0, 1);   // global_prefetch_b8
    v16bf bfrag;
#pragma unroll
    for (int e = 0; e < 16; ++e) bfrag[e] = f2bf(xb[(size_t)e * NPIX]);
#pragma unroll
    for (int mt = 0; mt < 12; ++mt) {
      const v16bf* ap = (const v16bf*)(aw + (size_t)(mt * KT + kt) * 512 + lane * 16);
      acc[mt] = wmma_bf16(*ap, bfrag, acc[mt]);
    }
  }

#pragma unroll
  for (int mt = 0; mt < 12; ++mt) {
#pragma unroll
    for (int j = 0; j < 8; ++j) {
      int m = (mBaseTile + mt) * 16 + j + 8 * kh;   // C layout: M = j (+8 hi half)
      float bv = bias ? bias[m] : 0.0f;
      Y[((size_t)b * M + m) * NPIX + n] = acc[mt][j] + bv;
    }
  }
}

// ---------------- KV projection with fused 2x2 Haar DWT ----------------------
// kv[b, g*96+o', y, x] = sum_{k'} kv_w[g*96+o', k'] * dwt(g*384+k')
__global__ void kv_dwt_kernel(const __bf16* __restrict__ Apack,
                              const float* __restrict__ x,
                              float* __restrict__ kv) {
  __shared__ __bf16 aw[6 * 12 * 512];           // 72 KB: this block's group
  const int g_blk = ((blockIdx.x * 8) >> 10) & 3;   // uniform per block
  stage_weights(aw, Apack + (size_t)g_blk * 6 * 12 * 512,
                (unsigned)(6 * 12 * 512 * 2));

  const int lane = threadIdx.x & 31;
  const int wave = blockIdx.x * (blockDim.x >> 5) + (threadIdx.x >> 5);
  const int ntile = wave & 1023;                // HPIX/16
  const int g     = (wave >> 10) & 3;
  const int b     = wave >> 12;
  const int col   = lane & 15;
  const int kh    = lane >> 4;
  const int n     = ntile * 16 + col;
  const int y     = n >> 7;
  const int xx    = n & 127;

  v8f zero = {0.f,0.f,0.f,0.f,0.f,0.f,0.f,0.f};
  v8f acc[6];
#pragma unroll
  for (int i = 0; i < 6; ++i) acc[i] = zero;

  for (int kt = 0; kt < 12; ++kt) {
    int idx0 = g * 384 + kt * 32 + kh * 16;     // multiple of 4
    int cx0 = idx0 >> 2;
    v16bf bfrag;
#pragma unroll
    for (int ci = 0; ci < 4; ++ci) {
      const float* xc = x + ((size_t)b * DIMC + (cx0 + ci)) * NPIX
                          + (size_t)(2 * y) * IMW + 2 * xx;
      float a  = xc[0];
      float bb = xc[1];
      float cc = xc[IMW];
      float dd = xc[IMW + 1];
      bfrag[4*ci + 0] = f2bf(0.5f * ( a + bb + cc + dd));   // ll
      bfrag[4*ci + 1] = f2bf(0.5f * (-a - bb + cc + dd));   // lh
      bfrag[4*ci + 2] = f2bf(0.5f * (-a + bb - cc + dd));   // hl
      bfrag[4*ci + 3] = f2bf(0.5f * ( a - bb - cc + dd));   // hh
    }
#pragma unroll
    for (int mt = 0; mt < 6; ++mt) {
      const v16bf* ap = (const v16bf*)(aw + (size_t)(mt * 12 + kt) * 512 + lane * 16);
      acc[mt] = wmma_bf16(*ap, bfrag, acc[mt]);
    }
  }

#pragma unroll
  for (int mt = 0; mt < 6; ++mt) {
#pragma unroll
    for (int j = 0; j < 8; ++j) {
      int o = g * 96 + mt * 16 + j + 8 * kh;
      kv[((size_t)b * DIMQ * 2 + o) * HPIX + n] = acc[mt][j];
    }
  }
}

// ---------------- Windowed linear attention (no softmax) ---------------------
// Per (batch, 8x8 q-window, head): corr = (q kT)*scale then attn = corr v.
// One wave per (window, head); LDS round-trip converts C-layout -> A-layout.
__global__ void attn_kernel(const float* __restrict__ qf,
                            const float* __restrict__ kvbuf,
                            float* __restrict__ afeat) {
  __shared__ float scorr[8][64 * 16];
  const int lane = threadIdx.x & 31;
  const int w    = threadIdx.x >> 5;
  const int wave = blockIdx.x * 8 + w;
  const int head = wave & 7;
  const int wx   = (wave >> 3) & 31;
  const int wy   = (wave >> 8) & 31;
  const int b    = wave >> 13;
  const int col  = lane & 15;
  const int kh   = lane >> 4;

  v8f zero = {0.f,0.f,0.f,0.f,0.f,0.f,0.f,0.f};

  // key position for this lane (4x4 window in the half-res map)
  const int ky = wy * 4 + (col >> 2);
  const int kx = wx * 4 + (col & 3);

  // ---- corr = q @ kT  (B-frag: B[d][key] = k_feat[key][d]) ----
  v16bf kfrag;
#pragma unroll
  for (int e = 0; e < 16; ++e) {
    int d = kh * 16 + e;
    float v = (d < HEADD)
        ? kvbuf[((size_t)b * (2*DIMQ) + head * HEADD + d) * HPIX + ky * 128 + kx]
        : 0.0f;
    kfrag[e] = f2bf(v);
  }

#pragma unroll
  for (int mt = 0; mt < 4; ++mt) {
    int q  = mt * 16 + col;
    int qy = wy * 8 + (q >> 3);
    int qx = wx * 8 + (q & 7);
    v16bf qfrag;
#pragma unroll
    for (int e = 0; e < 16; ++e) {
      int d = a_kidx(e, lane);
      float v = (d < HEADD)
          ? qf[((size_t)b * DIMQ + head * HEADD + d) * NPIX + qy * 256 + qx]
          : 0.0f;
      qfrag[e] = f2bf(v);
    }
    v8f cacc = wmma_bf16(qfrag, kfrag, zero);
#pragma unroll
    for (int j = 0; j < 8; ++j) {
      int qrow = mt * 16 + j + 8 * kh;
      scorr[w][qrow * 16 + col] = cacc[j] * SCALE_F;
    }
  }
  __syncthreads();

  // ---- attn = corr @ v  (keys padded 16->32, d split into 2 N-tiles) ----
  v16bf vfrag[2];
#pragma unroll
  for (int nt = 0; nt < 2; ++nt) {
    int d = nt * 16 + col;
#pragma unroll
    for (int e = 0; e < 16; ++e) {
      int key = kh * 16 + e;
      float v = (key < 16 && d < HEADD)
          ? kvbuf[((size_t)b * (2*DIMQ) + DIMQ + head * HEADD + d) * HPIX
                  + (wy * 4 + (key >> 2)) * 128 + (wx * 4 + (key & 3))]
          : 0.0f;
      vfrag[nt][e] = f2bf(v);
    }
  }

#pragma unroll
  for (int mt = 0; mt < 4; ++mt) {
    v16bf afr;
#pragma unroll
    for (int e = 0; e < 16; ++e) {
      int key = a_kidx(e, lane);
      afr[e] = (key < 16) ? f2bf(scorr[w][(mt * 16 + col) * 16 + key]) : f2bf(0.0f);
    }
#pragma unroll
    for (int nt = 0; nt < 2; ++nt) {
      v8f o = wmma_bf16(afr, vfrag[nt], zero);
#pragma unroll
      for (int j = 0; j < 8; ++j) {
        int q = mt * 16 + j + 8 * kh;
        int d = nt * 16 + col;
        if (d < HEADD) {
          int qy = wy * 8 + (q >> 3);
          int qx = wx * 8 + (q & 7);
          afeat[((size_t)b * DIMQ + head * HEADD + d) * NPIX + qy * 256 + qx] = o[j];
        }
      }
    }
  }
}

// ---------------- Launch --------------------------------------------------
extern "C" void kernel_launch(void* const* d_in, const int* in_sizes, int n_in,
                              void* d_out, int out_size, void* d_ws, size_t ws_size,
                              hipStream_t stream) {
  (void)in_sizes; (void)n_in; (void)out_size; (void)ws_size;
  const float* x      = (const float*)d_in[0];
  const float* q_w    = (const float*)d_in[1];
  const float* kv_w   = (const float*)d_in[2];
  const float* proj_w = (const float*)d_in[3];
  const float* proj_b = (const float*)d_in[4];
  float* out = (float*)d_out;

  // workspace carving (bytes)
  char* ws = (char*)d_ws;
  __bf16* qwP  = (__bf16*)(ws + 0);                    // 12*12*512*2  = 147456
  __bf16* kvP  = (__bf16*)(ws + 147456);               // 24*12*512*2  = 294912
  __bf16* pwP  = (__bf16*)(ws + 442368);               // 24*6*512*2   = 147456
  float*  qfeat = (float*)(ws + 589824);               // B*192*65536*4 = 201326592
  float*  kvbuf = qfeat + (size_t)BATCH * DIMQ * NPIX; // B*384*16384*4 = 100663296
  float*  afeat = kvbuf + (size_t)BATCH * 2 * DIMQ * HPIX;

  // 1) pack weights into bf16 WMMA A-fragment layout
  pack_w_kernel<<<(12*12*512 + 255) / 256, 256, 0, stream>>>(q_w,    qwP, DIMQ,   DIMC);
  pack_w_kernel<<<(24*12*512 + 255) / 256, 256, 0, stream>>>(kv_w,   kvP, 2*DIMQ, DIMC);
  pack_w_kernel<<<(24*6 *512 + 255) / 256, 256, 0, stream>>>(proj_w, pwP, DIMC,   DIMQ);

  // 2) q projection: 192x384 @ 384x65536 per batch
  gemm_chan_kernel<<<dim3(2048, 1), 256, 0, stream>>>(qwP, x, qfeat, nullptr, DIMQ, DIMC);

  // 3) kv projection with fused DWT: 4 groups of 96x384 @ 384x16384 per batch
  kv_dwt_kernel<<<2048, 256, 0, stream>>>(kvP, x, kvbuf);

  // 4) windowed linear attention -> afeat (B,192,256,256)
  attn_kernel<<<4096, 256, 0, stream>>>(qfeat, kvbuf, afeat);

  // 5) output projection + bias: 384x192 @ 192x65536 per batch (2 M-blocks)
  gemm_chan_kernel<<<dim3(2048, 2), 256, 0, stream>>>(pwP, afeat, out, proj_b, DIMC, DIMQ);
}
